// RSSM_57586921505250
// MI455X (gfx1250) — compile-verified, matrix operand order checked by reading, not run on hardware
//
#include <hip/hip_runtime.h>
#include <hip/hip_bf16.h>

// ---------------------------------------------------------------------------
// RSSM (DreamerV3-style) scan for MI455X / gfx1250.
// GEMMs: v_wmma_f32_16x16x32_bf16 with LDS double-buffering fed by
// global_load_async_to_lds_b128 (ASYNCcnt) -- CDNA5 async copy path.
// Weights converted once to transposed bf16 [N, Kpad]; activations staged bf16.
// ---------------------------------------------------------------------------

typedef __attribute__((ext_vector_type(16))) __bf16 v16bf;
typedef __attribute__((ext_vector_type(8)))  float  v8f;

#define B_     512
#define T_     64
#define E_     1536
#define A_     6
#define STOCH_ 32
#define DETER_ 512
#define HID_   512
#define OUTW_  704          // 6*STOCH + DETER
#define LN_EPS 1e-3f
#define MIN_STD 0.1f

// GEMM tiling
#define BM_ 128
#define BN_ 64
#define BK_ 64
#define PAD_ 8              // bf16 elements of padding (16B) per LDS row

__device__ __forceinline__ float sigm_(float x) { return 1.f / (1.f + expf(-x)); }
__device__ __forceinline__ float softplus_(float x) {
    return (x > 20.f) ? x : log1pf(expf(x));
}

// gfx1250 async copy: per-lane 16B global -> LDS, tracked by ASYNCcnt.
__device__ __forceinline__ void async_cp16_(unsigned lds_off, const __bf16* g) {
    asm volatile("global_load_async_to_lds_b128 %0, %1, off"
                 :: "v"(lds_off), "v"(g) : "memory");
}
__device__ __forceinline__ void wait_async6_() {
    asm volatile("s_wait_asynccnt 0x6" ::: "memory");
}
__device__ __forceinline__ void wait_async0_() {
    asm volatile("s_wait_asynccnt 0x0" ::: "memory");
}

// ---------------------------------------------------------------------------
// Weight convert: W [K,N] f32 row-major  ->  Wt [N,Kpad] bf16 (zero-padded K)
// ---------------------------------------------------------------------------
__global__ void k_cvt_w(const float* __restrict__ W, __bf16* __restrict__ Wt,
                        int K, int N, int Kpad) {
    long i = (long)blockIdx.x * blockDim.x + threadIdx.x;
    long total = (long)N * Kpad;
    if (i >= total) return;
    int n = (int)(i / Kpad);
    int k = (int)(i % Kpad);
    Wt[i] = (k < K) ? (__bf16)W[(long)k * N + n] : (__bf16)0.f;
}

// ---------------------------------------------------------------------------
// GEMM: C[M,N] f32 = A[M,K] bf16 x Bt[N,K] bf16^T.
// Requires M % 128 == 0, N % 64 == 0, K % 64 == 0.
// Grid: (N/BN, M/BM); 256 threads = 8 waves in 4(M) x 2(N); each wave 32x32.
// LDS double-buffered stages filled via async-to-LDS copies.
// ---------------------------------------------------------------------------
__global__ __launch_bounds__(256) void k_gemm_bf16(
        const __bf16* __restrict__ A, const __bf16* __restrict__ Bt,
        float* __restrict__ C, int M, int N, int K) {
    __shared__ __attribute__((aligned(16))) __bf16 As[2][BM_][BK_ + PAD_];
    __shared__ __attribute__((aligned(16))) __bf16 Bs[2][BN_][BK_ + PAD_];

    const int tid  = threadIdx.x;
    const int lane = tid & 31;
    const int wave = tid >> 5;
    const int bm = blockIdx.y * BM_;
    const int bn = blockIdx.x * BN_;
    const int wm = (wave >> 1) << 5;   // 0,32,64,96
    const int wn = (wave & 1) << 5;    // 0,32

    const int lr = lane & 15;          // row/col within 16-tile
    const int kh = (lane >> 4) << 4;   // K half (0 or 16)

    const int nstages = K / BK_;

    // issue one LDS stage: A tile 128x64 (1024 x 16B chunks, 4/thread),
    //                      B tile  64x64 ( 512 x 16B chunks, 2/thread)
    auto load_stage = [&](int s) {
        const int d  = s & 1;
        const int k0 = s * BK_;
        unsigned abase = (unsigned)(unsigned long long)&As[d][0][0];
        unsigned bbase = (unsigned)(unsigned long long)&Bs[d][0][0];
#pragma unroll
        for (int q = 0; q < 4; ++q) {
            int c = tid + q * 256;
            int row = c >> 3, col = (c & 7) * 8;
            async_cp16_(abase + (unsigned)(row * (BK_ + PAD_) + col) * 2u,
                        A + (size_t)(bm + row) * K + k0 + col);
        }
#pragma unroll
        for (int q = 0; q < 2; ++q) {
            int c = tid + q * 256;
            int row = c >> 3, col = (c & 7) * 8;
            async_cp16_(bbase + (unsigned)(row * (BK_ + PAD_) + col) * 2u,
                        Bt + (size_t)(bn + row) * K + k0 + col);
        }
    };

    v8f c00 = {}, c01 = {}, c10 = {}, c11 = {};

    load_stage(0);
    for (int s = 0; s < nstages; ++s) {
        if (s + 1 < nstages) { load_stage(s + 1); wait_async6_(); }
        else                 { wait_async0_(); }
        __syncthreads();
        const int d = s & 1;
#pragma unroll
        for (int kk = 0; kk < BK_; kk += 32) {
            v16bf fa0 = *(const v16bf*)&As[d][wm + lr][kk + kh];
            v16bf fa1 = *(const v16bf*)&As[d][wm + 16 + lr][kk + kh];
            v16bf fb0 = *(const v16bf*)&Bs[d][wn + lr][kk + kh];
            v16bf fb1 = *(const v16bf*)&Bs[d][wn + 16 + lr][kk + kh];
            c00 = __builtin_amdgcn_wmma_f32_16x16x32_bf16(false, fa0, false, fb0, (short)0, c00, false, false);
            c01 = __builtin_amdgcn_wmma_f32_16x16x32_bf16(false, fa0, false, fb1, (short)0, c01, false, false);
            c10 = __builtin_amdgcn_wmma_f32_16x16x32_bf16(false, fa1, false, fb0, (short)0, c10, false, false);
            c11 = __builtin_amdgcn_wmma_f32_16x16x32_bf16(false, fa1, false, fb1, (short)0, c11, false, false);
        }
        __syncthreads();
    }

    // C layout: VGPR r, lane L -> row = r + (L>=16 ? 8 : 0), col = L & 15
    const int rbase = (lane >> 4) * 8;
    const int col   = lane & 15;
    float* Cp = C + (size_t)(bm + wm + rbase) * N + bn + wn + col;
#pragma unroll
    for (int r = 0; r < 8; ++r) {
        Cp[(size_t)r * N]             = c00[r];
        Cp[(size_t)r * N + 16]        = c01[r];
        Cp[(size_t)(r + 16) * N]      = c10[r];
        Cp[(size_t)(r + 16) * N + 16] = c11[r];
    }
}

// ---------------------------------------------------------------------------
// Row-wise LayerNorm + SiLU + bf16 cast.  One block (512 thr) per row of 512.
// ---------------------------------------------------------------------------
__global__ void k_ln_silu(const float* __restrict__ pre,
                          const float* __restrict__ g, const float* __restrict__ bb,
                          __bf16* __restrict__ out, int out_stride) {
    __shared__ float s1[HID_], s2[HID_];
    int row = blockIdx.x, j = threadIdx.x;
    float x = pre[(size_t)row * HID_ + j];
    s1[j] = x; s2[j] = x * x;
    __syncthreads();
    for (int o = HID_ / 2; o > 0; o >>= 1) {
        if (j < o) { s1[j] += s1[j + o]; s2[j] += s2[j + o]; }
        __syncthreads();
    }
    float m = s1[0] * (1.f / HID_);
    float v = s2[0] * (1.f / HID_) - m * m;
    float y = (x - m) * rsqrtf(v + LN_EPS) * g[j] + bb[j];
    float s = y * sigm_(y);
    out[(size_t)row * out_stride + j] = (__bf16)s;
}

// ---------------------------------------------------------------------------
// Per-step prep: reset-blend deter (stage into A2 cols [512,1024)), build
// xin = bf16(concat(stoch_blend, action*(1-r), pad0)) [B, 64].
// ---------------------------------------------------------------------------
__global__ void k_prep(const float* __restrict__ action, const float* __restrict__ reset,
                       const float* __restrict__ stoch,
                       const float* __restrict__ deter0, const float* __restrict__ stoch0,
                       float* __restrict__ deter, __bf16* __restrict__ a2,
                       __bf16* __restrict__ xin, int t) {
    int bI = blockIdx.x, j = threadIdx.x;   // 512 threads
    float r = reset[(size_t)bI * T_ + t];
    float d = deter[(size_t)bI * DETER_ + j];
    d = d * (1.f - r) + deter0[j] * r;
    deter[(size_t)bI * DETER_ + j] = d;
    a2[(size_t)bI * 1024 + 512 + j] = (__bf16)d;
    if (j < 64) {
        float x;
        if (j < STOCH_) {
            float s0 = stoch[(size_t)bI * STOCH_ + j];
            x = s0 * (1.f - r) + stoch0[j] * r;
        } else if (j < STOCH_ + A_) {
            x = action[((size_t)bI * T_ + t) * A_ + (j - STOCH_)] * (1.f - r);
        } else {
            x = 0.f;
        }
        xin[(size_t)bI * 64 + j] = (__bf16)x;
    }
}

// ---------------------------------------------------------------------------
// GRU gate: parts[B,1536] -> deter_n; stage bf16 copies; write out deter slice.
// ---------------------------------------------------------------------------
__global__ void k_gru(const float* __restrict__ parts, float* __restrict__ deter,
                      __bf16* __restrict__ a_img, __bf16* __restrict__ a_obs,
                      float* __restrict__ out, int t) {
    int bI = blockIdx.x, j = threadIdx.x;   // 512 threads
    const float* p = parts + (size_t)bI * 1536;
    float rg   = sigm_(p[j]);
    float cand = tanhf(rg * p[512 + j]);
    float up   = sigm_(p[1024 + j] - 1.f);
    float d    = deter[(size_t)bI * DETER_ + j];
    float dn   = up * cand + (1.f - up) * d;
    deter[(size_t)bI * DETER_ + j] = dn;
    a_img[(size_t)bI * DETER_ + j] = (__bf16)dn;
    a_obs[(size_t)bI * 2048 + j]   = (__bf16)dn;
    out[((size_t)bI * T_ + t) * OUTW_ + 6 * STOCH_ + j] = dn;
}

// ---------------------------------------------------------------------------
// Stage embed[:, t, :] into a_obs cols [512, 2048) as bf16
// ---------------------------------------------------------------------------
__global__ void k_stage_embed(const float* __restrict__ embed,
                              __bf16* __restrict__ a_obs, int t) {
    long i = (long)blockIdx.x * blockDim.x + threadIdx.x;
    if (i >= (long)B_ * E_) return;
    int bI = (int)(i / E_);
    int k  = (int)(i % E_);
    a_obs[(size_t)bI * 2048 + 512 + k] =
        (__bf16)embed[((size_t)bI * T_ + t) * E_ + k];
}

// ---------------------------------------------------------------------------
// Dist stats + reparam sample.  st[B,64]; writes mean/std/sample to out slices.
// If stoch_next != nullptr, also stores sample as next carry stoch.
// ---------------------------------------------------------------------------
__global__ void k_stats(const float* __restrict__ st, const float* __restrict__ bias,
                        const float* __restrict__ noise, float* __restrict__ out,
                        float* __restrict__ stoch_next, int t, int out_off) {
    int bI = blockIdx.x, j = threadIdx.x;   // 32 threads
    float m    = st[(size_t)bI * 64 + j] + bias[j];
    float sraw = st[(size_t)bI * 64 + 32 + j] + bias[32 + j];
    float sd   = softplus_(sraw) + MIN_STD;
    float n    = noise[((size_t)t * B_ + bI) * STOCH_ + j];
    float smp  = m + sd * n;
    float* o = out + ((size_t)bI * T_ + t) * OUTW_ + out_off;
    o[j] = m; o[STOCH_ + j] = sd; o[2 * STOCH_ + j] = smp;
    if (stoch_next) stoch_next[(size_t)bI * STOCH_ + j] = smp;
}

// ---------------------------------------------------------------------------
// Learned initial state: deter0 = tanh(W_init); stoch0 = mean of prior(deter0)
// Single block, 512 threads.
// ---------------------------------------------------------------------------
__global__ void k_init_state(const float* __restrict__ W_init,
                             const float* __restrict__ W_img,
                             const float* __restrict__ g_img, const float* __restrict__ b_img,
                             const float* __restrict__ W_ims, const float* __restrict__ b_ims,
                             float* __restrict__ deter0, float* __restrict__ stoch0) {
    __shared__ float sh[HID_], red[HID_];
    int j = threadIdx.x;
    float d = tanhf(W_init[j]);
    deter0[j] = d;
    sh[j] = d;
    __syncthreads();
    float acc = 0.f;
    for (int k = 0; k < DETER_; ++k) acc += sh[k] * W_img[(size_t)k * HID_ + j];
    red[j] = acc; __syncthreads();
    for (int o = HID_ / 2; o > 0; o >>= 1) { if (j < o) red[j] += red[j + o]; __syncthreads(); }
    float m = red[0] * (1.f / HID_);
    __syncthreads();
    float c = acc - m;
    red[j] = c * c; __syncthreads();
    for (int o = HID_ / 2; o > 0; o >>= 1) { if (j < o) red[j] += red[j + o]; __syncthreads(); }
    float v = red[0] * (1.f / HID_);
    __syncthreads();
    float y = c * rsqrtf(v + LN_EPS) * g_img[j] + b_img[j];
    float s = y * sigm_(y);
    sh[j] = s;
    __syncthreads();
    if (j < STOCH_) {
        float a2 = b_ims[j];
        for (int k = 0; k < HID_; ++k) a2 += sh[k] * W_ims[(size_t)k * 64 + j];
        stoch0[j] = a2;   // mean -> mode of prior dist
    }
}

__global__ void k_broadcast(const float* __restrict__ deter0, const float* __restrict__ stoch0,
                            float* __restrict__ deter, float* __restrict__ stoch) {
    int bI = blockIdx.x, j = threadIdx.x;   // 512 threads
    deter[(size_t)bI * DETER_ + j] = deter0[j];
    if (j < STOCH_) stoch[(size_t)bI * STOCH_ + j] = stoch0[j];
}

// ---------------------------------------------------------------------------
extern "C" void kernel_launch(void* const* d_in, const int* in_sizes, int n_in,
                              void* d_out, int out_size, void* d_ws, size_t ws_size,
                              hipStream_t stream) {
    const float* embed       = (const float*)d_in[0];
    const float* action      = (const float*)d_in[1];
    const float* reset       = (const float*)d_in[2];
    const float* noise_post  = (const float*)d_in[3];
    const float* noise_prior = (const float*)d_in[4];
    const float* W_inp  = (const float*)d_in[5];
    const float* g_inp  = (const float*)d_in[6];
    const float* b_inp  = (const float*)d_in[7];
    const float* W_gru  = (const float*)d_in[8];
    const float* W_img  = (const float*)d_in[9];
    const float* g_img  = (const float*)d_in[10];
    const float* b_img  = (const float*)d_in[11];
    const float* W_obs  = (const float*)d_in[12];
    const float* g_obs  = (const float*)d_in[13];
    const float* b_obs  = (const float*)d_in[14];
    const float* W_ims  = (const float*)d_in[15];
    const float* b_ims  = (const float*)d_in[16];
    const float* W_ostat = (const float*)d_in[17];
    const float* b_ostat = (const float*)d_in[18];
    const float* W_init = (const float*)d_in[19];
    float* out = (float*)d_out;

    // ---- workspace bump allocator (256B aligned) ----
    char* ws = (char*)d_ws;
    size_t off = 0;
    auto alloc = [&](size_t bytes) -> char* {
        char* p = ws + off;
        off = (off + bytes + 255) & ~(size_t)255;
        return p;
    };
    __bf16* Wt_inp   = (__bf16*)alloc((size_t)512 * 64 * 2);
    __bf16* Wt_gru   = (__bf16*)alloc((size_t)1536 * 1024 * 2);
    __bf16* Wt_img   = (__bf16*)alloc((size_t)512 * 512 * 2);
    __bf16* Wt_obs   = (__bf16*)alloc((size_t)512 * 2048 * 2);
    __bf16* Wt_ims   = (__bf16*)alloc((size_t)64 * 512 * 2);
    __bf16* Wt_ostat = (__bf16*)alloc((size_t)64 * 512 * 2);
    __bf16* xin   = (__bf16*)alloc((size_t)B_ * 64 * 2);
    __bf16* A2    = (__bf16*)alloc((size_t)B_ * 1024 * 2);
    __bf16* a_img = (__bf16*)alloc((size_t)B_ * 512 * 2);
    __bf16* hi    = (__bf16*)alloc((size_t)B_ * 512 * 2);
    __bf16* a_obs = (__bf16*)alloc((size_t)B_ * 2048 * 2);
    __bf16* hh    = (__bf16*)alloc((size_t)B_ * 512 * 2);
    float* pre_x  = (float*)alloc((size_t)B_ * 512 * 4);
    float* parts  = (float*)alloc((size_t)B_ * 1536 * 4);
    float* pre_hi = (float*)alloc((size_t)B_ * 512 * 4);
    float* pre_h  = (float*)alloc((size_t)B_ * 512 * 4);
    float* st_p   = (float*)alloc((size_t)B_ * 64 * 4);
    float* st_q   = (float*)alloc((size_t)B_ * 64 * 4);
    float* deter  = (float*)alloc((size_t)B_ * DETER_ * 4);
    float* stoch  = (float*)alloc((size_t)B_ * STOCH_ * 4);
    float* deter0 = (float*)alloc((size_t)DETER_ * 4);
    float* stoch0 = (float*)alloc((size_t)STOCH_ * 4);
    (void)ws_size; (void)in_sizes; (void)n_in; (void)out_size;

    auto cvt = [&](const float* W, __bf16* Wt, int K, int N, int Kpad) {
        long total = (long)N * Kpad;
        k_cvt_w<<<(unsigned)((total + 255) / 256), 256, 0, stream>>>(W, Wt, K, N, Kpad);
    };
    cvt(W_inp,   Wt_inp,   STOCH_ + A_, HID_, 64);
    cvt(W_gru,   Wt_gru,   1024, 1536, 1024);
    cvt(W_img,   Wt_img,   DETER_, HID_, 512);
    cvt(W_obs,   Wt_obs,   DETER_ + E_, HID_, 2048);
    cvt(W_ims,   Wt_ims,   HID_, 64, 512);
    cvt(W_ostat, Wt_ostat, HID_, 64, 512);

    k_init_state<<<1, 512, 0, stream>>>(W_init, W_img, g_img, b_img, W_ims, b_ims,
                                        deter0, stoch0);
    k_broadcast<<<B_, 512, 0, stream>>>(deter0, stoch0, deter, stoch);

    auto gemm = [&](const __bf16* A, const __bf16* Bt, float* C, int M, int N, int K) {
        dim3 grid((unsigned)(N / BN_), (unsigned)(M / BM_));
        k_gemm_bf16<<<grid, 256, 0, stream>>>(A, Bt, C, M, N, K);
    };

    for (int t = 0; t < T_; ++t) {
        // --- img_step: input MLP ---
        k_prep<<<B_, 512, 0, stream>>>(action, reset, stoch, deter0, stoch0,
                                       deter, A2, xin, t);
        gemm(xin, Wt_inp, pre_x, B_, HID_, 64);
        k_ln_silu<<<B_, 512, 0, stream>>>(pre_x, g_inp, b_inp, A2, 1024);
        // --- GRU ---
        gemm(A2, Wt_gru, parts, B_, 1536, 1024);
        k_gru<<<B_, 512, 0, stream>>>(parts, deter, a_img, a_obs, out, t);
        // --- prior: img_out + ims_stats + sample ---
        k_stage_embed<<<(unsigned)(((long)B_ * E_ + 255) / 256), 256, 0, stream>>>(embed, a_obs, t);
        gemm(a_img, Wt_img, pre_hi, B_, HID_, 512);
        k_ln_silu<<<B_, 512, 0, stream>>>(pre_hi, g_img, b_img, hi, 512);
        gemm(hi, Wt_ims, st_p, B_, 64, 512);
        k_stats<<<B_, STOCH_, 0, stream>>>(st_p, b_ims, noise_prior, out,
                                           nullptr, t, 3 * STOCH_);
        // --- posterior ---
        gemm(a_obs, Wt_obs, pre_h, B_, HID_, 2048);
        k_ln_silu<<<B_, 512, 0, stream>>>(pre_h, g_obs, b_obs, hh, 512);
        gemm(hh, Wt_ostat, st_q, B_, 64, 512);
        k_stats<<<B_, STOCH_, 0, stream>>>(st_q, b_ostat, noise_post, out,
                                           stoch, t, 0);
    }
}
